// DualCrossAttend_53008486367819
// MI455X (gfx1250) — matrix operand order repaired and checked by reading, not compiled, not run on hardware
//
#include <hip/hip_runtime.h>
#include <hip/hip_bf16.h>
#include <math.h>

// ---------------------------------------------------------------------------
// DualCrossAttend fused pipeline for MI455X (gfx1250, wave32, WMMA bf16)
//   stage 0 (pack kernels): fp32 weights -> bf16, WMMA B-fragment-major
//   stage 1 (fused kernel): 32 rows / workgroup, 512 threads = 16 waves.
//   Each wave computes a 32(M)x16(N) tile = two wmma accumulators sharing one
//   B fragment -> halves L2 weight traffic vs a 16-row tile.
//   Inputs are streamed with non-temporal loads (read-once, 336 MB) so the
//   packed weights (~1.2 MB) stay resident in the 192 MB L2.
// ---------------------------------------------------------------------------

typedef __bf16 bf16_t;
typedef bf16_t v16bf __attribute__((ext_vector_type(16)));
typedef float  v8f   __attribute__((ext_vector_type(8)));
typedef float  f32x4 __attribute__((ext_vector_type(4)));
typedef unsigned int u32;
typedef u32 u32x4 __attribute__((ext_vector_type(4)));

union Frag { u32 u[8]; u32x4 q[2]; v16bf v; };

__device__ __forceinline__ unsigned short f2bf(float f) {
  union { float f; u32 u; } a; a.f = f;
  u32 r = a.u + 0x7fffu + ((a.u >> 16) & 1u);   // round-to-nearest-even
  return (unsigned short)(r >> 16);
}
__device__ __forceinline__ float bf2f(unsigned short b) {
  union { u32 u; float f; } a; a.u = ((u32)b) << 16;
  return a.f;
}
__device__ __forceinline__ u32 pk2(float lo, float hi) {
  return (u32)f2bf(lo) | ((u32)f2bf(hi) << 16);
}

// ---------------------------------------------------------------------------
// Weight pack: W[K][N] fp32 (row stride ld, column offset col0) ->
// fragment-major packed bf16 pairs for V_WMMA_F32_16X16X32_BF16 B operand.
// Word index = ((kb*(N/16) + nb)*32 + lane)*8 + v
//   lane: n = nb*16 + (lane&15), h = lane>>4
//   word v holds bf16 pair (K = 32*kb + 16*(v>>2) + 8*h + 2*(v&3) , K+1) at column n.
// A wave's B fragment is then 8 consecutive dwords -> 2x global_load_b128.
// ---------------------------------------------------------------------------
__global__ void pack_weight_kernel(const float* __restrict__ src, int ld, int col0,
                                   int K, int N, u32* __restrict__ dst) {
  int t = blockIdx.x * blockDim.x + threadIdx.x;
  int total = (K >> 1) * N;
  if (t >= total) return;
  int NB = N >> 4;
  int v  = t & 7;
  int L  = (t >> 3) & 31;
  int nb = (t >> 8) % NB;
  int kb = (t >> 8) / NB;
  int h  = L >> 4;
  int n  = nb * 16 + (L & 15);
  int k  = kb * 32 + 16 * (v >> 2) + 8 * h + 2 * (v & 3);
  float lo = src[(size_t)k * ld + col0 + n];
  float hi = src[(size_t)(k + 1) * ld + col0 + n];
  dst[t] = pk2(lo, hi);
}

// ---------------------------------------------------------------------------
// 32(M) x 16(N) tile, K = 32*kIters. A bf16 row-major in LDS (aStrideWords
// dwords/row, 32 rows), B fragment-major packed in global. One shared B
// fragment feeds two wmma (M rows 0-15 and 16-31) per K step.
// 16-bit A fragment layout (ISA 7.12.2): lanes 0-15 (h=0) -> K {0..7,16..23},
// lanes 16-31 (h=1) -> K {8..15,24..31}; pairs are consecutive -> b128 loads.
// ---------------------------------------------------------------------------
__device__ __forceinline__ void tile_gemm2(const u32* __restrict__ A, int aStrideWords,
                                           const u32* __restrict__ Bpk, int NB, int nb,
                                           int kIters, int lane, v8f& c0, v8f& c1) {
  const int h = lane >> 4;
  const int m = lane & 15;
  const u32* a0 = A + (size_t)m * aStrideWords + 4 * h;
  const u32* a1 = a0 + 16 * (size_t)aStrideWords;
  const u32* bp = Bpk + (((size_t)nb) * 32 + lane) * 8;
  const size_t bStep = (size_t)NB * 32 * 8;
  for (int kb = 0; kb < kIters; ++kb) {
    Frag a, b, a2;
    const int kw = kb * 16;
    b.q[0]  = *(const u32x4*)(bp);            // global_load_b128 (shared B)
    b.q[1]  = *(const u32x4*)(bp + 4);
    a.q[0]  = *(const u32x4*)(a0 + kw);       // ds_load_b128, rows 0..15
    a.q[1]  = *(const u32x4*)(a0 + kw + 8);
    a2.q[0] = *(const u32x4*)(a1 + kw);       // ds_load_b128, rows 16..31
    a2.q[1] = *(const u32x4*)(a1 + kw + 8);
    bp += bStep;
    c0 = __builtin_amdgcn_wmma_f32_16x16x32_bf16(false, a.v,  false, b.v,
                                                 (short)0, c0, false, false);
    c1 = __builtin_amdgcn_wmma_f32_16x16x32_bf16(false, a2.v, false, b.v,
                                                 (short)0, c1, false, false);
  }
}

#define ROWS 32
#define SMEM_BYTES 155648

__global__ __launch_bounds__(512, 1)
void fused_dual_cross_attend(const float* __restrict__ Xg, const float* __restrict__ Xl,
                             const u32* __restrict__ WgP, const float* __restrict__ bg,
                             const u32* __restrict__ WlP, const float* __restrict__ bl,
                             const u32* __restrict__ WqP, const u32* __restrict__ WkP,
                             const u32* __restrict__ WvP, const float* __restrict__ b_in,
                             const u32* __restrict__ WoP, const float* __restrict__ b_out,
                             const float* __restrict__ gamma, const float* __restrict__ beta,
                             const u32* __restrict__ W1P, const float* __restrict__ b1,
                             const float* __restrict__ W2,  const float* __restrict__ b2,
                             float* __restrict__ out) {
  extern __shared__ __attribute__((aligned(16))) unsigned char smem[];
  u32* sIn = (u32*)(smem);              // 32 x 320 dwords (32x640 bf16)  40960 B
  u32* sGb = (u32*)(smem + 40960);      // 32 x 256 bf16                  16384 B
  u32* sLb = (u32*)(smem + 57344);      // 32 x 256 bf16                  16384 B
  u32* sQb = (u32*)(smem + 73728);      // 32 x 256 bf16                  16384 B
  u32* sKb = (u32*)(smem + 90112);      // 32 x 512 bf16 (tokens g|l)     32768 B
  u32* sVb = (u32*)(smem + 122880);     // 32 x 512 bf16                  32768 B
  // aliases (lifetimes do not overlap with original contents):
  unsigned short* sCtx = (unsigned short*)(smem);          // 32x256 bf16 over sIn
  unsigned short* sYb  = (unsigned short*)(smem + 16384);  // 32x256 bf16 over sIn
  unsigned short* sHb  = (unsigned short*)(smem + 32768);  // 32x128 bf16 over sIn
  float*          sXf  = (float*)(smem + 90112);           // 32x256 f32 over sKb

  const int t    = threadIdx.x;
  const int lane = t & 31;
  const int wv   = t >> 5;
  const int h    = lane >> 4;
  const int mr   = lane & 15;
  const int row0 = blockIdx.x * ROWS;

  // ---- phase A: stream global_diff tile (non-temporal), fp32->bf16 in LDS --
  for (int i = t; i < ROWS * 160; i += 512) {
    int r = i / 160, c = i % 160;
    f32x4 f = __builtin_nontemporal_load((const f32x4*)(Xg + (size_t)(row0 + r) * 640) + c);
    sIn[r * 320 + c * 2]     = pk2(f.x, f.y);
    sIn[r * 320 + c * 2 + 1] = pk2(f.z, f.w);
  }
  __syncthreads();

  // ---- g = Xg @ Wg + bg : 16 waves x (32x16 tile), K=640 -------------------
  {
    v8f c0 = {0.f,0.f,0.f,0.f,0.f,0.f,0.f,0.f}, c1 = c0;
    tile_gemm2(sIn, 320, WgP, 16, wv, 20, lane, c0, c1);
    int n = wv * 16 + mr;
    float bias = bg[n];
    unsigned short* gb = (unsigned short*)sGb;
    #pragma unroll
    for (int r = 0; r < 8; ++r) {
      int m = r + 8 * h;
      gb[m * 256 + n]        = f2bf(c0[r] + bias);
      gb[(m + 16) * 256 + n] = f2bf(c1[r] + bias);
    }
  }
  __syncthreads();

  // ---- phase B: stream local_diff tile (reuse sIn) -------------------------
  for (int i = t; i < ROWS * 160; i += 512) {
    int r = i / 160, c = i % 160;
    f32x4 f = __builtin_nontemporal_load((const f32x4*)(Xl + (size_t)(row0 + r) * 640) + c);
    sIn[r * 320 + c * 2]     = pk2(f.x, f.y);
    sIn[r * 320 + c * 2 + 1] = pk2(f.z, f.w);
  }
  __syncthreads();

  // ---- l = Xl @ Wl + bl ----------------------------------------------------
  {
    v8f c0 = {0.f,0.f,0.f,0.f,0.f,0.f,0.f,0.f}, c1 = c0;
    tile_gemm2(sIn, 320, WlP, 16, wv, 20, lane, c0, c1);
    int n = wv * 16 + mr;
    float bias = bl[n];
    unsigned short* lb = (unsigned short*)sLb;
    #pragma unroll
    for (int r = 0; r < 8; ++r) {
      int m = r + 8 * h;
      lb[m * 256 + n]        = f2bf(c0[r] + bias);
      lb[(m + 16) * 256 + n] = f2bf(c1[r] + bias);
    }
  }
  __syncthreads();

  // ---- QKV: q=g@Wq, k={g,l}@Wk, v={g,l}@Wv (K=256) -------------------------
  for (int gm = 0; gm < 5; ++gm) {
    const u32* A  = (gm == 2 || gm == 4) ? sLb : sGb;
    const u32* Bw = (gm == 0) ? WqP : (gm < 3) ? WkP : WvP;
    int boff      = (gm == 0) ? 0 : (gm < 3) ? 256 : 512;
    v8f c0 = {0.f,0.f,0.f,0.f,0.f,0.f,0.f,0.f}, c1 = c0;
    tile_gemm2(A, 128, Bw, 16, wv, 8, lane, c0, c1);
    int n = wv * 16 + mr;
    float bias = b_in[boff + n];
    unsigned short* dst;
    int stride, col;
    if      (gm == 0) { dst = (unsigned short*)sQb; stride = 256; col = n;       }
    else if (gm == 1) { dst = (unsigned short*)sKb; stride = 512; col = n;       }
    else if (gm == 2) { dst = (unsigned short*)sKb; stride = 512; col = 256 + n; }
    else if (gm == 3) { dst = (unsigned short*)sVb; stride = 512; col = n;       }
    else              { dst = (unsigned short*)sVb; stride = 512; col = 256 + n; }
    #pragma unroll
    for (int r = 0; r < 8; ++r) {
      int m = r + 8 * h;
      dst[m * stride + col]        = f2bf(c0[r] + bias);
      dst[(m + 16) * stride + col] = f2bf(c1[r] + bias);
    }
  }
  __syncthreads();

  // ---- attention: 2-key softmax per (row, head), 128 scalar threads --------
  if (t < 128) {
    int r = t >> 2, hd = t & 3;
    const unsigned short* q  = (const unsigned short*)sQb + r * 256 + hd * 64;
    const unsigned short* kg = (const unsigned short*)sKb + r * 512 + hd * 64;
    const unsigned short* kl = kg + 256;
    const unsigned short* vg = (const unsigned short*)sVb + r * 512 + hd * 64;
    const unsigned short* vl = vg + 256;
    float s0 = 0.f, s1 = 0.f;
    for (int d = 0; d < 64; ++d) {
      float qd = bf2f(q[d]);
      s0 += qd * bf2f(kg[d]);
      s1 += qd * bf2f(kl[d]);
    }
    s0 *= 0.125f; s1 *= 0.125f;                 // 1/sqrt(64)
    float mx = fmaxf(s0, s1);
    float e0 = __expf(s0 - mx), e1 = __expf(s1 - mx);
    float inv = 1.f / (e0 + e1);
    float a0 = e0 * inv, a1 = e1 * inv;
    for (int d = 0; d < 64; ++d)
      sCtx[r * 256 + hd * 64 + d] = f2bf(a0 * bf2f(vg[d]) + a1 * bf2f(vl[d]));
  }
  __syncthreads();

  // ---- attended = ctx @ W_out + b_out ; x = g + attended -------------------
  {
    v8f c0 = {0.f,0.f,0.f,0.f,0.f,0.f,0.f,0.f}, c1 = c0;
    tile_gemm2((const u32*)sCtx, 128, WoP, 16, wv, 8, lane, c0, c1);
    int n = wv * 16 + mr;
    float bias = b_out[n];
    const unsigned short* gb = (const unsigned short*)sGb;
    #pragma unroll
    for (int r = 0; r < 8; ++r) {
      int m = r + 8 * h;
      sXf[m * 256 + n]        = c0[r] + bias + bf2f(gb[m * 256 + n]);
      sXf[(m + 16) * 256 + n] = c1[r] + bias + bf2f(gb[(m + 16) * 256 + n]);
    }
  }
  __syncthreads();

  // ---- LayerNorm(256): 16 threads per row (xor-reduce within half-wave) ----
  {
    int r = t >> 4, sub = t & 15;
    const float* x = sXf + r * 256;
    float s = 0.f, sq = 0.f;
    #pragma unroll
    for (int i = 0; i < 16; ++i) {
      float v = x[sub * 16 + i];
      s += v; sq += v * v;
    }
    for (int o = 8; o; o >>= 1) {
      s  += __shfl_xor(s,  o, 32);
      sq += __shfl_xor(sq, o, 32);
    }
    float mu  = s * (1.f / 256.f);
    float var = sq * (1.f / 256.f) - mu * mu;
    float inv = rsqrtf(var + 1e-5f);
    #pragma unroll
    for (int i = 0; i < 16; ++i) {
      int n = sub * 16 + i;
      sYb[r * 256 + n] = f2bf((x[n] - mu) * inv * gamma[n] + beta[n]);
    }
  }
  __syncthreads();

  // ---- h = exact_gelu(y @ W1 + b1) : 8 N-tiles (waves 0..7) ----------------
  if (wv < 8) {
    v8f c0 = {0.f,0.f,0.f,0.f,0.f,0.f,0.f,0.f}, c1 = c0;
    tile_gemm2((const u32*)sYb, 128, W1P, 8, wv, 8, lane, c0, c1);
    int n = wv * 16 + mr;
    float bias = b1[n];
    #pragma unroll
    for (int r = 0; r < 8; ++r) {
      int m = r + 8 * h;
      float x0 = c0[r] + bias, x1 = c1[r] + bias;
      sHb[m * 128 + n]        = f2bf(0.5f * x0 * (1.f + erff(x0 * 0.70710678118654752f)));
      sHb[(m + 16) * 128 + n] = f2bf(0.5f * x1 * (1.f + erff(x1 * 0.70710678118654752f)));
    }
  }
  __syncthreads();

  // ---- out = h @ W2 + b2 : 16 threads per row ------------------------------
  {
    int r = t >> 4, sub = t & 15;
    float s = 0.f;
    #pragma unroll
    for (int i = 0; i < 8; ++i) {
      int n = sub * 8 + i;
      s += bf2f(sHb[r * 128 + n]) * W2[n];
    }
    for (int o = 8; o; o >>= 1) s += __shfl_xor(s, o, 32);
    if (sub == 0) out[row0 + r] = s + b2[0];
  }
}

// ---------------------------------------------------------------------------
extern "C" void kernel_launch(void* const* d_in, const int* in_sizes, int n_in,
                              void* d_out, int out_size, void* d_ws, size_t ws_size,
                              hipStream_t stream) {
  const float* Xg    = (const float*)d_in[0];
  const float* Xl    = (const float*)d_in[1];
  const float* Wg    = (const float*)d_in[2];
  const float* bg    = (const float*)d_in[3];
  const float* Wl    = (const float*)d_in[4];
  const float* bl    = (const float*)d_in[5];
  const float* W_in  = (const float*)d_in[6];
  const float* b_in  = (const float*)d_in[7];
  const float* W_out = (const float*)d_in[8];
  const float* b_out = (const float*)d_in[9];
  const float* gamma = (const float*)d_in[10];
  const float* beta  = (const float*)d_in[11];
  const float* W1    = (const float*)d_in[12];
  const float* b1    = (const float*)d_in[13];
  const float* W2    = (const float*)d_in[14];
  const float* b2    = (const float*)d_in[15];

  // packed-weight workspace layout (dwords); total 311296 dwords ~= 1.19 MB
  u32* ws  = (u32*)d_ws;
  u32* WgP = ws;            // 640x256 -> 81920
  u32* WlP = ws + 81920;    // 640x256 -> 81920
  u32* WqP = ws + 163840;   // 256x256 -> 32768
  u32* WkP = ws + 196608;   // 256x256 -> 32768
  u32* WvP = ws + 229376;   // 256x256 -> 32768
  u32* WoP = ws + 262144;   // 256x256 -> 32768
  u32* W1P = ws + 294912;   // 256x128 -> 16384

  auto pack = [&](const float* src, int ld, int col0, int K, int N, u32* dst) {
    int words = (K >> 1) * N;
    pack_weight_kernel<<<(words + 255) / 256, 256, 0, stream>>>(src, ld, col0, K, N, dst);
  };
  pack(Wg,    256,   0, 640, 256, WgP);
  pack(Wl,    256,   0, 640, 256, WlP);
  pack(W_in,  768,   0, 256, 256, WqP);   // q | k | v columns of packed in-proj
  pack(W_in,  768, 256, 256, 256, WkP);
  pack(W_in,  768, 512, 256, 256, WvP);
  pack(W_out, 256,   0, 256, 256, WoP);
  pack(W1,    128,   0, 256, 128, W1P);

  int B = in_sizes[0] / 640;              // 65536
  dim3 grid(B / ROWS), block(512);
  fused_dual_cross_attend<<<grid, block, SMEM_BYTES, stream>>>(
      Xg, Xl, WgP, bg, WlP, bl, WqP, WkP, WvP, b_in,
      WoP, b_out, gamma, beta, W1P, b1, W2, b2, (float*)d_out);
}
